// SupervisedContrastiveLoss_81028853006775
// MI455X (gfx1250) — matrix operand order repaired and checked by reading, not compiled, not run on hardware
//
#include <hip/hip_runtime.h>
#include <hip/hip_bf16.h>

#define INV_T      (1.0f / 0.07f)
#define TEMP_RATIO (0.07f / 0.07f)   // TEMPERATURE / BASE_TEMPERATURE
#define JSPLIT     4                 // column-range split for occupancy

typedef __attribute__((ext_vector_type(8)))  float          v8f;
typedef __attribute__((ext_vector_type(16))) __bf16         v16bf;
typedef __attribute__((ext_vector_type(8)))  unsigned short u16x8;
typedef __attribute__((ext_vector_type(16))) unsigned short u16x16;

union FragU {
  struct H { u16x8 lo; u16x8 hi; } h;
  u16x16 v;
  v16bf  b;
};

__device__ __forceinline__ unsigned short f32_to_bf16_rn(float f) {
  unsigned int u = __float_as_uint(f);
  u += 0x7FFFu + ((u >> 16) & 1u);           // round-to-nearest-even
  return (unsigned short)(u >> 16);
}
__device__ __forceinline__ float bf16_to_f32(unsigned short h) {
  return __uint_as_float(((unsigned int)h) << 16);
}
__device__ __forceinline__ v16bf load_frag(const unsigned short* p0,
                                           const unsigned short* p1) {
  FragU f;
  f.h.lo = *(const u16x8*)p0;   // 16B aligned
  f.h.hi = *(const u16x8*)p1;   // 16B aligned
  return f.b;
}

// ---------------- setup kernels ----------------

// zero per-row partial-sum arrays (atomically accumulated), histogram, scalars
__global__ void zero_kernel(float* sumE, float* sumP, float* diag,
                            int* hist, float* acc, int N) {
  int i = blockIdx.x * blockDim.x + threadIdx.x;
  if (i < N) { sumE[i] = 0.0f; sumP[i] = 0.0f; diag[i] = 0.0f; }
  if (i < 256) hist[i] = 0;
  if (i < 2)   acc[i]  = 0.0f;
}

// group label per row + histogram of group labels
__global__ void prep_kernel(const int* __restrict__ labels, const int* __restrict__ kp,
                            int* __restrict__ glab, int* __restrict__ hist, int N) {
  int i = blockIdx.x * blockDim.x + threadIdx.x;
  if (i >= N) return;
  int k = kp[0]; if (k < 1) k = 1;
  int lb = labels[(i / k) * k];
  glab[i] = lb;
  if ((i % k) == 0 && lb >= 0 && lb < 256) atomicAdd(&hist[lb], 1);
}

// one wave per row: L2-normalize, split into bf16 hi + lo
__global__ void normalize_kernel(const float* __restrict__ X,
                                 unsigned short* __restrict__ fhi,
                                 unsigned short* __restrict__ flo, int N, int D) {
  int wave = threadIdx.x >> 5;
  int lane = threadIdx.x & 31;
  int row  = blockIdx.x * (blockDim.x >> 5) + wave;
  if (row >= N) return;
  int per = D >> 5;                              // 16 for D=512
  const float* xr = X + (size_t)row * D + lane * per;
  float vals[16];
  float ss = 0.0f;
  for (int i = 0; i < per; ++i) { float v = xr[i]; vals[i] = v; ss += v * v; }
  #pragma unroll
  for (int m = 16; m >= 1; m >>= 1) ss += __shfl_xor(ss, m, 32);
  float inv = 1.0f / fmaxf(sqrtf(ss), 1e-12f);
  unsigned short* ph = fhi + (size_t)row * D + lane * per;
  unsigned short* pl = flo + (size_t)row * D + lane * per;
  for (int i = 0; i < per; ++i) {
    float xn = vals[i] * inv;
    unsigned short h = f32_to_bf16_rn(xn);
    unsigned short l = f32_to_bf16_rn(xn - bf16_to_f32(h));
    ph[i] = h; pl[i] = l;
  }
}

// ---------------- fused GEMM + softmax statistics ----------------
// grid = (N/64, JSPLIT); block = 128 threads = 4 waves; wave owns a 16-row
// tile and a 1/JSPLIT slice of the column tiles (stats are additive, combined
// with atomics). K in chunks of 32 with split-bf16 WMMA, JB=4 column tiles
// per A-fragment load.
__global__ __launch_bounds__(128)
void sim_kernel(const unsigned short* __restrict__ fhi,
                const unsigned short* __restrict__ flo,
                const int* __restrict__ glab,
                float* __restrict__ sumE_out, float* __restrict__ sumP_out,
                float* __restrict__ diag_out, int N, int D) {
  const int wave = threadIdx.x >> 5;
  const int lane = threadIdx.x & 31;
  const int sub  = lane >> 4;          // half-wave selector
  const int nidx = lane & 15;
  const int i0   = (blockIdx.x * 4 + wave) * 16;

  const int JT    = N >> 4;                        // total 16-col tiles
  const int jtLen = JT / JSPLIT;
  const int jtBeg = blockIdx.y * jtLen;
  const int jtEnd = jtBeg + jtLen;

  // A fragment addressing (contiguous k = 16*sub + s, same order as B)
  const int aRow = i0 + nidx;
  const unsigned short* aHiBase = fhi + (size_t)aRow * D;
  const unsigned short* aLoBase = flo + (size_t)aRow * D;
  const int aOff1 = sub * 16;
  const int aOff2 = sub * 16 + 8;

  // C layout: lane holds rows i0 + 8*sub + r at column j0 + nidx
  int cRow[8]; int rowLab[8];
  #pragma unroll
  for (int r = 0; r < 8; ++r) { cRow[r] = i0 + sub * 8 + r; rowLab[r] = glab[cRow[r]]; }

  float sumE[8], sumP[8], dg[8];
  #pragma unroll
  for (int r = 0; r < 8; ++r) { sumE[r] = 0.0f; sumP[r] = 0.0f; dg[r] = 0.0f; }

  const int KC = D >> 5;     // 16 K-chunks of 32

  for (int jt = jtBeg; jt < jtEnd; jt += 4) {
    v8f acc[4] = {};
    for (int kc = 0; kc < KC; ++kc) {
      const int k0 = kc * 32;
      v16bf Ahi = load_frag(aHiBase + k0 + aOff1, aHiBase + k0 + aOff2);
      v16bf Alo = load_frag(aLoBase + k0 + aOff1, aLoBase + k0 + aOff2);
      #pragma unroll
      for (int jj = 0; jj < 4; ++jj) {
        const int bRow = (jt + jj) * 16 + nidx;           // B 32x16 layout
        const unsigned short* bH = fhi + (size_t)bRow * D + k0 + sub * 16;
        const unsigned short* bL = flo + (size_t)bRow * D + k0 + sub * 16;
        v16bf Bhi = load_frag(bH, bH + 8);
        v16bf Blo = load_frag(bL, bL + 8);
        // split-fp32 product: hi*hi + hi*lo + lo*hi (lo*lo negligible)
        acc[jj] = __builtin_amdgcn_wmma_f32_16x16x32_bf16(false, Ahi, false, Bhi,
                                                          (short)0, acc[jj], false, false);
        acc[jj] = __builtin_amdgcn_wmma_f32_16x16x32_bf16(false, Ahi, false, Blo,
                                                          (short)0, acc[jj], false, false);
        acc[jj] = __builtin_amdgcn_wmma_f32_16x16x32_bf16(false, Alo, false, Bhi,
                                                          (short)0, acc[jj], false, false);
      }
    }
    // softmax statistics: shift by the analytic row max (diag == 1.0)
    #pragma unroll
    for (int jj = 0; jj < 4; ++jj) {
      const int col = (jt + jj) * 16 + nidx;
      const int cl  = glab[col];
      #pragma unroll
      for (int r = 0; r < 8; ++r) {
        float v = acc[jj][r];
        sumE[r] += __expf((v - 1.0f) * INV_T);
        bool isDiag = (col == cRow[r]);
        bool pos    = (cl == rowLab[r]) && !isDiag;
        sumP[r] += pos ? v : 0.0f;
        if (isDiag) dg[r] = v;        // exactly one lane/split per row hits this
      }
    }
  }

  // reduce across the 16 lanes of each half-wave (xor 1,2,4,8 stays in-half),
  // then combine partial column-slices with atomics
  #pragma unroll
  for (int r = 0; r < 8; ++r) {
    float e = sumE[r], p = sumP[r], d = dg[r];
    #pragma unroll
    for (int m = 1; m <= 8; m <<= 1) {
      e += __shfl_xor(e, m, 32);
      p += __shfl_xor(p, m, 32);
      d += __shfl_xor(d, m, 32);
    }
    if (nidx == 0) {
      int row = cRow[r];
      atomicAdd(&sumE_out[row], e);
      atomicAdd(&sumP_out[row], p);
      atomicAdd(&diag_out[row], d);   // non-owning splits add 0.0
    }
  }
}

// ---------------- per-row loss + global reduction ----------------

__global__ void finalize_kernel(const float* __restrict__ sumE,
                                const float* __restrict__ sumP,
                                const float* __restrict__ diag,
                                const int* __restrict__ glab,
                                const int* __restrict__ hist,
                                const int* __restrict__ kp,
                                float* __restrict__ acc, int N) {
  int i = blockIdx.x * blockDim.x + threadIdx.x;
  if (i >= N) return;
  int k = kp[0]; if (k < 1) k = 1;
  int g = glab[i]; if (g < 0 || g > 255) g = 0;
  float cnt = (float)(hist[g] * k - 1);                       // positives per row
  float denom = sumE[i] - __expf((diag[i] - 1.0f) * INV_T) + 1e-8f;
  // sum over positives of (logits - M - log denom), M = 1/T (analytic row max)
  float mlpp = (sumP[i] * INV_T - cnt * (INV_T + logf(denom))) / fmaxf(cnt, 1.0f);
  float per  = -TEMP_RATIO * mlpp;
  bool valid = cnt > 0.0f;
  atomicAdd(&acc[0], valid ? per : 0.0f);
  atomicAdd(&acc[1], valid ? 1.0f : 0.0f);
}

__global__ void final_kernel(const float* __restrict__ acc, float* __restrict__ out) {
  out[0] = acc[0] / fmaxf(acc[1], 1.0f);
}

// ---------------- host-side launch ----------------

extern "C" void kernel_launch(void* const* d_in, const int* in_sizes, int n_in,
                              void* d_out, int out_size, void* d_ws, size_t ws_size,
                              hipStream_t stream) {
  const float* X      = (const float*)d_in[0];
  const int*   labels = (const int*)d_in[1];
  const int*   kp     = (const int*)d_in[2];   // rotation_multiplier (device scalar)
  const int N = in_sizes[1];                   // 8192
  const int D = in_sizes[0] / N;               // 512

  // workspace layout (~16.13 MB)
  char* ws = (char*)d_ws;
  size_t off = 0;
  unsigned short* fhi = (unsigned short*)(ws + off); off += (size_t)N * D * 2;
  unsigned short* flo = (unsigned short*)(ws + off); off += (size_t)N * D * 2;
  float* sumE = (float*)(ws + off); off += (size_t)N * 4;
  float* sumP = (float*)(ws + off); off += (size_t)N * 4;
  float* diag = (float*)(ws + off); off += (size_t)N * 4;
  int*   glab = (int*)(ws + off);   off += (size_t)N * 4;
  int*   hist = (int*)(ws + off);   off += 256 * 4;
  float* acc  = (float*)(ws + off); off += 2 * 4;
  (void)ws_size; (void)n_in; (void)out_size;

  zero_kernel<<<(N + 255) / 256, 256, 0, stream>>>(sumE, sumP, diag, hist, acc, N);
  prep_kernel<<<(N + 255) / 256, 256, 0, stream>>>(labels, kp, glab, hist, N);
  normalize_kernel<<<(N + 7) / 8, 256, 0, stream>>>(X, fhi, flo, N, D);
  dim3 gridSim(N / 64, JSPLIT);
  sim_kernel<<<gridSim, 128, 0, stream>>>(fhi, flo, glab, sumE, sumP, diag, N, D);
  finalize_kernel<<<(N + 255) / 256, 256, 0, stream>>>(sumE, sumP, diag, glab, hist, kp, acc, N);
  final_kernel<<<1, 1, 0, stream>>>(acc, d_out ? (float*)d_out : acc);
}
